// FixedSplitModel_34583076667906
// MI455X (gfx1250) — matrix operand order repaired and checked by reading, not compiled, not run on hardware
//
#include <hip/hip_runtime.h>
#include <hip/hip_bf16.h>

#define Bn    256
#define Ln    1024
#define Hn    64
#define H2n   128
#define HALFn 32
#define Vn    64
#define NTOK  (Bn * Ln)          // 262144 tokens
#define NTILE (NTOK / 16)        // 16384 tiles of 16 tokens
#define NBLK  1024
#define WPB   2                  // waves per block
#define GW    (NBLK * WPB)       // 2048 global waves -> 8 tiles each

typedef _Float16 v16h __attribute__((ext_vector_type(16)));
typedef _Float16 v8h  __attribute__((ext_vector_type(8)));
typedef float    v8f  __attribute__((ext_vector_type(8)));
typedef float    v4f  __attribute__((ext_vector_type(4)));

// ---- WMMA helper (CDNA5 v_wmma_f32_16x16x32_f16, wave32) --------------------

__device__ __forceinline__ v8f wmma_f16(v16h a, v16h b, v8f c) {
    return __builtin_amdgcn_wmma_f32_16x16x32_f16(false, a, false, b,
                                                  (short)0, c, false, false);
}

__device__ __forceinline__ v16h join8(v8h lo, v8h hi) {
    return __builtin_shufflevector(lo, hi, 0, 1, 2, 3, 4, 5, 6, 7,
                                   8, 9, 10, 11, 12, 13, 14, 15);
}

// A fragment from a row-major contiguous f16 row: two aligned b128 LDS loads.
// Element i: K = kbase + (i<8 ? 0 : 16) + h*8 + (i&7)  (h = lane>>4)
__device__ __forceinline__ v16h fragArow(const _Float16* rowp, int h, int kbase) {
    v8h lo = *(const v8h*)(rowp + kbase + h * 8);
    v8h hi = *(const v8h*)(rowp + kbase + 16 + h * 8);
    return join8(lo, hi);
}

// B fragment from pre-swizzled fragment-order storage: two b128 LDS loads.
__device__ __forceinline__ v16h fragBpre(const _Float16* base, int frag, int lane) {
    const v8h* p = (const v8h*)(base + ((size_t)(frag * 32 + lane)) * 16);
    return join8(p[0], p[1]);
}

// Stage weight W [Kd x Nd] (row-major f32) into fragment order:
// dst[((c*T + t)*32 + lane)*16 + i], lane = h*16 + nl,
// where k = 32c + (i<8 ? 0 : 16) + 8h + (i&7), n = 16t + nl.
template <int Kd, int Nd>
__device__ __forceinline__ void stageB(_Float16* dst, const float* src,
                                       int tid, int nthreads) {
    constexpr int T = Nd >> 4;
    for (int idx = tid; idx < Kd * Nd; idx += nthreads) {
        int k = idx / Nd, n = idx % Nd;           // Nd power of two -> shifts
        int c = k >> 5, kr = k & 31;
        int h = (kr >> 3) & 1, i = (kr & 7) | ((kr >> 4) << 3);
        int t = n >> 4, nl = n & 15;
        dst[(((c * T + t) * 32) + (h * 16 + nl)) * 16 + i] = (_Float16)src[idx];
    }
}

// ---- Kernel 1: embed-gather + FFN + residual + LayerNorm + projections ------

__global__ __launch_bounds__(WPB * 32)
void fsm_encoder_kernel(const int* __restrict__ seq,
                        const float* __restrict__ embed,
                        const float* __restrict__ W1, const float* __restrict__ b1,
                        const float* __restrict__ W2, const float* __restrict__ b2,
                        const float* __restrict__ gamma, const float* __restrict__ beta,
                        const float* __restrict__ Wsem, const float* __restrict__ bsem,
                        const float* __restrict__ Wepi, const float* __restrict__ bepi,
                        float* __restrict__ ksw, float* __restrict__ kew) {
    __shared__ __align__(16) _Float16 sEmb[Vn * Hn];     // row-major (A + residual)
    __shared__ __align__(16) _Float16 sW1[Hn * H2n];     // fragment order
    __shared__ __align__(16) _Float16 sW2[H2n * Hn];     // fragment order
    __shared__ __align__(16) _Float16 sWs[Hn * HALFn];   // fragment order
    __shared__ __align__(16) _Float16 sWe[Hn * HALFn];   // fragment order
    __shared__ float sB1[H2n], sB2[Hn], sBs[HALFn], sBe[HALFn], sG[Hn], sBt[Hn];
    __shared__ __align__(16) _Float16 sT1[WPB][16 * H2n];  // per-wave staging
    __shared__ __align__(16) _Float16 sH [WPB][16 * Hn];
    __shared__ int sIds[WPB][16];

    const int tid  = threadIdx.x;
    const int wave = tid >> 5;
    const int lane = tid & 31;
    const int nl   = lane & 15;
    const int h    = lane >> 4;
    const int NT   = WPB * 32;

    for (int i = tid; i < Vn * Hn; i += NT) sEmb[i] = (_Float16)embed[i];
    stageB<Hn, H2n>(sW1, W1, tid, NT);
    stageB<H2n, Hn>(sW2, W2, tid, NT);
    stageB<Hn, HALFn>(sWs, Wsem, tid, NT);
    stageB<Hn, HALFn>(sWe, Wepi, tid, NT);
    for (int i = tid; i < H2n;   i += NT) sB1[i] = b1[i];
    for (int i = tid; i < Hn;    i += NT) { sB2[i] = b2[i]; sG[i] = gamma[i]; sBt[i] = beta[i]; }
    for (int i = tid; i < HALFn; i += NT) { sBs[i] = bsem[i]; sBe[i] = bepi[i]; }
    __syncthreads();

    const int gw = blockIdx.x * WPB + wave;    // NTILE % GW == 0 -> uniform trips
    for (int tile = gw; tile < NTILE; tile += GW) {
        const long tok0 = (long)tile * 16;
        const int  bIdx = (int)(tok0 >> 10);   // tiles never cross a batch (1024%16==0)
        const int  l0   = (int)(tok0 & (Ln - 1));

        if (lane < 16) sIds[wave][lane] = seq[tok0 + lane];
        __syncthreads();

        const _Float16* erow = sEmb + sIds[wave][nl] * Hn;

        // ---- GEMM1: [16x64] @ W1[64x128], +b1, relu -> sT1 ----
        v8f acc1[8] = {};
#pragma unroll
        for (int c = 0; c < 2; ++c) {
            v16h a = fragArow(erow, h, 32 * c);
#pragma unroll
            for (int t = 0; t < 8; ++t)
                acc1[t] = wmma_f16(a, fragBpre(sW1, c * 8 + t, lane), acc1[t]);
        }
#pragma unroll
        for (int t = 0; t < 8; ++t) {
#pragma unroll
            for (int r = 0; r < 8; ++r) {
                int mm = r + 8 * h, nn = 16 * t + nl;
                float v = acc1[t][r] + sB1[nn];
                sT1[wave][mm * H2n + nn] = (_Float16)(v > 0.f ? v : 0.f);
            }
        }
        __syncthreads();

        // ---- GEMM2: [16x128] @ W2[128x64] + b2 + residual ----
        v8f acc2[4] = {};
        const _Float16* t1row = sT1[wave] + nl * H2n;
#pragma unroll
        for (int c = 0; c < 4; ++c) {
            v16h a = fragArow(t1row, h, 32 * c);
#pragma unroll
            for (int t = 0; t < 4; ++t)
                acc2[t] = wmma_f16(a, fragBpre(sW2, c * 4 + t, lane), acc2[t]);
        }
#pragma unroll
        for (int t = 0; t < 4; ++t) {
#pragma unroll
            for (int r = 0; r < 8; ++r) {
                int mm = r + 8 * h, nn = 16 * t + nl;
                acc2[t][r] += sB2[nn] + (float)sEmb[sIds[wave][mm] * Hn + nn];
            }
        }

        // ---- LayerNorm over H=64 ----
#pragma unroll
        for (int r = 0; r < 8; ++r) {
            float s = acc2[0][r] + acc2[1][r] + acc2[2][r] + acc2[3][r];
            s += __shfl_xor(s, 1, 16); s += __shfl_xor(s, 2, 16);
            s += __shfl_xor(s, 4, 16); s += __shfl_xor(s, 8, 16);
            float mu = s * (1.0f / 64.0f);
            float q = acc2[0][r] * acc2[0][r] + acc2[1][r] * acc2[1][r] +
                      acc2[2][r] * acc2[2][r] + acc2[3][r] * acc2[3][r];
            q += __shfl_xor(q, 1, 16); q += __shfl_xor(q, 2, 16);
            q += __shfl_xor(q, 4, 16); q += __shfl_xor(q, 8, 16);
            float var = q * (1.0f / 64.0f) - mu * mu;
            float rs = rsqrtf(var + 1e-5f);
            int mm = r + 8 * h;
#pragma unroll
            for (int t = 0; t < 4; ++t) {
                int nn = 16 * t + nl;
                sH[wave][mm * Hn + nn] =
                    (_Float16)(sG[nn] * (acc2[t][r] - mu) * rs + sBt[nn]);
            }
        }
        __syncthreads();

        // ---- projections: h @ Wsem, h @ Wepi  ([16x64]@[64x32]) ----
        v8f accS[2] = {}, accE[2] = {};
        const _Float16* hrow = sH[wave] + nl * Hn;
#pragma unroll
        for (int c = 0; c < 2; ++c) {
            v16h a = fragArow(hrow, h, 32 * c);
#pragma unroll
            for (int t = 0; t < 2; ++t) {
                accS[t] = wmma_f16(a, fragBpre(sWs, c * 2 + t, lane), accS[t]);
                accE[t] = wmma_f16(a, fragBpre(sWe, c * 2 + t, lane), accE[t]);
            }
        }
        // transposed workspace layout: ks[b][half][l]  (scan streams along l)
#pragma unroll
        for (int t = 0; t < 2; ++t) {
#pragma unroll
            for (int r = 0; r < 8; ++r) {
                int mm = r + 8 * h, nn = 16 * t + nl;
                size_t idx = ((size_t)bIdx * HALFn + nn) * Ln + (l0 + mm);
                ksw[idx] = accS[t][r] + sBs[nn];
                kew[idx] = accE[t][r] + sBe[nn];
            }
        }
    }
}

// ---- Kernel 2: per-batch delta-rule scan + readout --------------------------

__device__ __forceinline__ float lanebcast(float v, int j) {
    return __int_as_float(__builtin_amdgcn_readlane(__float_as_int(v), j));
}

// One rank-1 delta update.  Matvec/dot use split accumulators so the serial
// dependence chain is ~7 deep instead of 32 (the scan is latency-bound).
__device__ __forceinline__ void delta_step(float* M, float kv, float scale) {
    float kr[32];
#pragma unroll
    for (int j = 0; j < 32; ++j) kr[j] = lanebcast(kv, j);

    float dp[4] = {0.f, 0.f, 0.f, 0.f};
#pragma unroll
    for (int j = 0; j < 32; ++j) dp[j & 3] += kr[j] * kr[j];
    float dot = (dp[0] + dp[1]) + (dp[2] + dp[3]);

    float pa[8] = {0.f, 0.f, 0.f, 0.f, 0.f, 0.f, 0.f, 0.f};
#pragma unroll
    for (int j = 0; j < 32; ++j) pa[j & 7] += M[j] * kr[j];
    float acc = ((pa[0] + pa[1]) + (pa[2] + pa[3])) +
                ((pa[4] + pa[5]) + (pa[6] + pa[7]));

    float dv = (kv - acc / (dot + 1e-6f)) * scale;
#pragma unroll
    for (int j = 0; j < 32; ++j) M[j] += dv * kr[j];
}

__global__ __launch_bounds__(32)
void fsm_scan_kernel(const float* __restrict__ ksw, const float* __restrict__ kew,
                     const float* __restrict__ Wout, const float* __restrict__ bout,
                     float* __restrict__ out) {
    const int b = blockIdx.x;
    const int lane = threadIdx.x;            // lane i owns row i of M_s / M_e
    // lane-contiguous streams: ks[b][lane][0..L)
    const float* ksb = ksw + ((size_t)b * HALFn + lane) * Ln;
    const float* keb = kew + ((size_t)b * HALFn + lane) * Ln;

    float ms[32], me[32];
#pragma unroll
    for (int j = 0; j < 32; ++j) { ms[j] = 0.f; me[j] = 0.f; }

    // main loop: 4 timesteps per b128 load; t = 0 .. 1019
    for (int t = 0; t < Ln - 4; t += 4) {
        v4f kv = *(const v4f*)(ksb + t);
        v4f ev = *(const v4f*)(keb + t);
        __builtin_prefetch(ksb + t + 16, 0, 0);
        __builtin_prefetch(keb + t + 16, 0, 0);
#pragma unroll
        for (int u = 0; u < 4; ++u) {
            delta_step(ms, kv[u], 1.0f);
            delta_step(me, ev[u], (float)(t + u + 1) * (1.0f / (float)Ln));
        }
    }
    // tail: t = 1020..1022 update, element 3 is the final key
    v4f kv = *(const v4f*)(ksb + (Ln - 4));
    v4f ev = *(const v4f*)(keb + (Ln - 4));
#pragma unroll
    for (int u = 0; u < 3; ++u) {
        delta_step(ms, kv[u], 1.0f);
        delta_step(me, ev[u], (float)(Ln - 4 + u + 1) * (1.0f / (float)Ln));
    }

    // final contractions + [1x64] @ Wout[64x64] + bout
    float cs4[4] = {0.f, 0.f, 0.f, 0.f}, ce4[4] = {0.f, 0.f, 0.f, 0.f};
#pragma unroll
    for (int j = 0; j < 32; ++j) cs4[j & 3] += ms[j] * lanebcast(kv[3], j);
#pragma unroll
    for (int j = 0; j < 32; ++j) ce4[j & 3] += me[j] * lanebcast(ev[3], j);
    float cs = (cs4[0] + cs4[1]) + (cs4[2] + cs4[3]);
    float ce = (ce4[0] + ce4[1]) + (ce4[2] + ce4[3]);

    __shared__ float cbuf[2 * HALFn];
    cbuf[lane] = cs;
    cbuf[HALFn + lane] = ce;
    __syncthreads();

#pragma unroll
    for (int vv = 0; vv < 2; ++vv) {
        int v = lane + 32 * vv;
        float o = bout[v];
        float p0 = 0.f, p1 = 0.f;
#pragma unroll 8
        for (int i = 0; i < HALFn; ++i) {
            p0 += cbuf[i] * Wout[i * Vn + v];
            p1 += cbuf[HALFn + i] * Wout[(HALFn + i) * Vn + v];
        }
        out[(size_t)b * Vn + v] = o + p0 + p1;
    }
}

extern "C" void kernel_launch(void* const* d_in, const int* in_sizes, int n_in,
                              void* d_out, int out_size, void* d_ws, size_t ws_size,
                              hipStream_t stream) {
    const int*   seq   = (const int*)  d_in[0];
    const float* embed = (const float*)d_in[1];
    const float* W1    = (const float*)d_in[2];
    const float* b1    = (const float*)d_in[3];
    const float* W2    = (const float*)d_in[4];
    const float* b2    = (const float*)d_in[5];
    const float* gamma = (const float*)d_in[6];
    const float* beta  = (const float*)d_in[7];
    const float* Wsem  = (const float*)d_in[8];
    const float* bsem  = (const float*)d_in[9];
    const float* Wepi  = (const float*)d_in[10];
    const float* bepi  = (const float*)d_in[11];
    const float* Wout  = (const float*)d_in[12];
    const float* bout  = (const float*)d_in[13];
    float* out = (float*)d_out;

    float* ksw = (float*)d_ws;                           // [B,32,L]
    float* kew = ksw + (size_t)NTOK * HALFn;             // [B,32,L]

    fsm_encoder_kernel<<<NBLK, WPB * 32, 0, stream>>>(
        seq, embed, W1, b1, W2, b2, gamma, beta, Wsem, bsem, Wepi, bepi, ksw, kew);
    fsm_scan_kernel<<<Bn, 32, 0, stream>>>(ksw, kew, Wout, bout, out);
}
